// My_rnn_79353815761337
// MI455X (gfx1250) — matrix operand order repaired and checked by reading, not compiled, not run on hardware
//
#include <hip/hip_runtime.h>
#include <utility>

// Problem dims (from reference): B=128, T=256, F=64, U=1024, OUT=32, DHID=32
#define B_DIM 128
#define T_DIM 256
#define F_DIM 64
#define U_DIM 1024
#define NOUT  32
#define DHID  32

typedef __attribute__((ext_vector_type(16))) __bf16 v16bf;
typedef __attribute__((ext_vector_type(8)))  __bf16 v8bf;
typedef __attribute__((ext_vector_type(8)))  float  v8f;

__device__ __forceinline__ float sigmoidf_(float x) {
    return 1.0f / (1.0f + __expf(-x));
}

// CDNA5 async global->LDS copy (16B per active lane), tracked with ASYNCcnt.
// VDST = 32-bit LDS byte offset (low 32 bits of a generic shared pointer),
// VADDR = 64-bit global address.
__device__ __forceinline__ void async_b128(void* lds, const void* gaddr) {
    uint32_t l = (uint32_t)(uintptr_t)lds;
    asm volatile("global_load_async_to_lds_b128 %0, %1, off"
                 :: "v"(l), "v"(gaddr) : "memory");
}
__device__ __forceinline__ void wait_async0() {
#if __has_builtin(__builtin_amdgcn_s_wait_asynccnt)
    __builtin_amdgcn_s_wait_asynccnt(0);
#else
    asm volatile("s_wait_asynccnt 0x0" ::: "memory");
#endif
}

// dst[n*pitch + k] = bf16(src[k*N + n])  — transpose + downconvert so a WMMA
// B-fragment (one output column, contiguous K) is two 16B loads per lane.
__global__ __launch_bounds__(256)
void transpose_to_bf16(__bf16* __restrict__ dst, int pitch,
                       const float* __restrict__ src, int K, int N)
{
    int i = blockIdx.x * 256 + threadIdx.x;
    if (i >= K * N) return;
    int n = i / K;
    int k = i - n * K;
    dst[(size_t)n * pitch + k] = (__bf16)src[(size_t)k * N + n];
}

__global__ __launch_bounds__(256)
void cast_to_bf16(__bf16* __restrict__ dst, const float* __restrict__ src, int count)
{
    int i = blockIdx.x * 256 + threadIdx.x;
    if (i < count) dst[i] = (__bf16)src[i];
}

__global__ __launch_bounds__(256)
void zero_buf(float* __restrict__ pf, __bf16* __restrict__ pb, int count)
{
    int i = blockIdx.x * 256 + threadIdx.x;
    if (i < count) { pf[i] = 0.0f; pb[i] = (__bf16)0.0f; }
}

// One fused LSTM step: z = [h|x] @ WT^T + b ; gates ; h_out(bf16)/c_out(f32).
//   KT = U + Kx (1088 for encoder / lstm_cell, 2048 for rnn_cell)
//   WT: [4U][KT] bf16 (row = output gate-column, contiguous K = [R-col | W-col])
// Block: 256 thr = 8 waves. Block tile: 16 batch rows x 64 U-cols.
// Wave w: gate g = w>>1 (Keras order i,f,g,o), col pair ct = w&1; each wave
// computes TWO 16x16 subtiles sharing one A-fragment (halves A LDS reads).
// Grid: (U/64, B/16) = (16, 8).
template<int KT>
__global__ __launch_bounds__(256)
void lstm_step_kernel(const __bf16* __restrict__ h_in,   // [B][U] bf16
                      const float*  __restrict__ c_in,   // [B][U] f32
                      const __bf16* __restrict__ x_in, int x_stride,  // bf16
                      const __bf16* __restrict__ WT,
                      const float*  __restrict__ bias,   // [4U]
                      __bf16* __restrict__ h_out,
                      float*  __restrict__ c_out)
{
    constexpr int KX    = KT - U_DIM;
    constexpr int PITCH = KT + 8;          // bf16 units; 16B-aligned rows, breaks bank stride
    extern __shared__ char smem[];
    __bf16* sA = (__bf16*)smem;            // [16][PITCH] staged bf16 [h|x] rows
    float*  sZ = (float*)smem;             // aliases sA after K-loop: [16][16][16] z tiles

    const int j0  = blockIdx.x * 64;       // U-column tile base
    const int m0  = blockIdx.y * 16;       // batch row base
    const int tid = threadIdx.x;

    // Async-stage A = concat(h, x): pure bf16 byte copies, no VGPR round-trip.
    constexpr int HCH = U_DIM / 8;         // 16B chunks per h row
    for (int c = tid; c < 16 * HCH; c += 256) {
        int r = c / HCH, q = c - r * HCH;
        async_b128((void*)(sA + r * PITCH + q * 8),
                   (const void*)(h_in + (size_t)(m0 + r) * U_DIM + q * 8));
    }
    constexpr int XCH = KX / 8;            // 16B chunks per x row
    for (int c = tid; c < 16 * XCH; c += 256) {
        int r = c / XCH, q = c - r * XCH;
        async_b128((void*)(sA + r * PITCH + U_DIM + q * 8),
                   (const void*)(x_in + (size_t)(m0 + r) * x_stride + q * 8));
    }
    wait_async0();
    __syncthreads();

    const int wave = tid >> 5;
    const int lane = tid & 31;
    const int gate = wave >> 1;            // 0:i 1:f 2:g 3:o
    const int ct   = wave & 1;
    const int n    = lane & 15;
    const int half = lane >> 4;
    const int gcol0 = gate * U_DIM + j0 + ct * 32 + n;   // subtile 0 column
    // subtile 1 column = gcol0 + 16

    // ISA 16-bit A layout: lane<16 -> K {k0..k0+7, k0+16..k0+23}; lane>=16 -> +8
    const __bf16* aptr  = sA + n * PITCH + half * 8;
    // B layout matches A: lane<16 -> col n, K {k0..k0+7, k0+16..k0+23}; lane>=16 -> +8
    const __bf16* bptr0 = WT + (size_t)gcol0 * KT + half * 8;
    const __bf16* bptr1 = bptr0 + (size_t)16 * KT;

    v8f acc0 = {}, acc1 = {};
    #pragma unroll 4
    for (int k0 = 0; k0 < KT; k0 += 32) {
        v8bf a0 = *(const v8bf*)(aptr + k0);
        v8bf a1 = *(const v8bf*)(aptr + k0 + 16);
        v16bf a = __builtin_shufflevector(a0, a1, 0,1,2,3,4,5,6,7,8,9,10,11,12,13,14,15);
        v16bf b0 = __builtin_shufflevector(*(const v8bf*)(bptr0 + k0), *(const v8bf*)(bptr0 + k0 + 16), 0,1,2,3,4,5,6,7,8,9,10,11,12,13,14,15);
        v16bf b1 = __builtin_shufflevector(*(const v8bf*)(bptr1 + k0), *(const v8bf*)(bptr1 + k0 + 16), 0,1,2,3,4,5,6,7,8,9,10,11,12,13,14,15);
        acc0 = __builtin_amdgcn_wmma_f32_16x16x32_bf16(false, a, false, b0,
                                                       (short)0, acc0, false, false);
        acc1 = __builtin_amdgcn_wmma_f32_16x16x32_bf16(false, a, false, b1,
                                                       (short)0, acc1, false, false);
    }

    float bv0 = bias[gcol0];
    float bv1 = bias[gcol0 + 16];
    __syncthreads();   // everyone done reading sA before z aliases it
    const int t0 = gate * 4 + ct * 2;      // z tile index (gate-major, then jj>>4)
    #pragma unroll
    for (int r = 0; r < 8; ++r) {          // D layout: lane<16 rows 0..7, lane>=16 rows 8..15
        sZ[(t0    ) * 256 + (half * 8 + r) * 16 + n] = acc0[r] + bv0;
        sZ[(t0 + 1) * 256 + (half * 8 + r) * 16 + n] = acc1[r] + bv1;
    }
    __syncthreads();

    // Gate math: c' = sig(f)*c + sig(i)*tanh(g); h' = sig(o)*tanh(c')
    for (int e = tid; e < 1024; e += 256) {
        int row = e >> 6;
        int jj  = e & 63;
        int c2  = jj >> 4;
        int nn  = jj & 15;
        float zi = sZ[(0 * 4 + c2) * 256 + row * 16 + nn];
        float zf = sZ[(1 * 4 + c2) * 256 + row * 16 + nn];
        float zg = sZ[(2 * 4 + c2) * 256 + row * 16 + nn];
        float zo = sZ[(3 * 4 + c2) * 256 + row * 16 + nn];
        size_t gb = (size_t)(m0 + row) * U_DIM + j0 + jj;
        float cprev = c_in[gb];
        float cn = sigmoidf_(zf) * cprev + sigmoidf_(zi) * tanhf(zg);
        float hn = sigmoidf_(zo) * tanhf(cn);
        c_out[gb] = cn;
        h_out[gb] = (__bf16)hn;
    }
}

// Dense head: pred = relu(h@W1 + b1) @ W2 + b2, all via WMMA.
// Block: 128 thr = 4 waves, 16 batch rows. Grid: B/16 = 8.
__global__ __launch_bounds__(128)
void dense_head_kernel(const __bf16* __restrict__ h_in,  // [B][U] bf16
                       const __bf16* __restrict__ W1T,   // [DHID][U]
                       const float*  __restrict__ b1,    // [DHID]
                       const __bf16* __restrict__ W2T,   // [F][DHID]
                       const float*  __restrict__ b2,    // [F]
                       float*  __restrict__ out, int out_stride,  // f32 into d_out
                       __bf16* __restrict__ pred)        // [B][F] bf16 feedback
{
    constexpr int PITCH = U_DIM + 8;
    __shared__ __bf16 sA[16 * PITCH];
    __shared__ __bf16 sH[16 * 40];       // hidden 16x32 bf16, pitch 40 (16B-aligned rows)
    const int m0  = blockIdx.x * 16;
    const int tid = threadIdx.x;

    constexpr int HCH = U_DIM / 8;
    for (int c = tid; c < 16 * HCH; c += 128) {
        int r = c >> 7, q = c & 127;
        async_b128((void*)(sA + r * PITCH + q * 8),
                   (const void*)(h_in + (size_t)(m0 + r) * U_DIM + q * 8));
    }
    wait_async0();
    __syncthreads();

    const int wave = tid >> 5;
    const int lane = tid & 31;
    const int n    = lane & 15;
    const int half = lane >> 4;

    if (wave < 2) {   // phase 1: hidden = relu(h@W1+b1), cols wave*16..+16, K=1024
        const __bf16* aptr = sA + n * PITCH + half * 8;
        const int col = wave * 16 + n;
        const __bf16* bptr = W1T + (size_t)col * U_DIM + half * 8;
        v8f acc = {};
        #pragma unroll 4
        for (int k0 = 0; k0 < U_DIM; k0 += 32) {
            v8bf a0 = *(const v8bf*)(aptr + k0);
            v8bf a1 = *(const v8bf*)(aptr + k0 + 16);
            v16bf a = __builtin_shufflevector(a0, a1, 0,1,2,3,4,5,6,7,8,9,10,11,12,13,14,15);
            v16bf b = __builtin_shufflevector(*(const v8bf*)(bptr + k0), *(const v8bf*)(bptr + k0 + 16), 0,1,2,3,4,5,6,7,8,9,10,11,12,13,14,15);
            acc = __builtin_amdgcn_wmma_f32_16x16x32_bf16(false, a, false, b,
                                                          (short)0, acc, false, false);
        }
        float bv = b1[col];
        #pragma unroll
        for (int r = 0; r < 8; ++r) {
            float v = fmaxf(acc[r] + bv, 0.0f);
            sH[(half * 8 + r) * 40 + col] = (__bf16)v;
        }
    }
    __syncthreads();

    {   // phase 2: pred = hidden@W2 + b2, cols wave*16..+16, K=32 (one WMMA)
        const __bf16* aptr = sH + n * 40 + half * 8;
        v8bf a0 = *(const v8bf*)(aptr);
        v8bf a1 = *(const v8bf*)(aptr + 16);
        v16bf a = __builtin_shufflevector(a0, a1, 0,1,2,3,4,5,6,7,8,9,10,11,12,13,14,15);
        const int col = wave * 16 + n;
        v16bf b = __builtin_shufflevector(*(const v8bf*)(W2T + (size_t)col * DHID + half * 8), *(const v8bf*)(W2T + (size_t)col * DHID + half * 8 + 16), 0,1,2,3,4,5,6,7,8,9,10,11,12,13,14,15);
        v8f acc = {};
        acc = __builtin_amdgcn_wmma_f32_16x16x32_bf16(false, a, false, b,
                                                      (short)0, acc, false, false);
        float bv = b2[col];
        #pragma unroll
        for (int r = 0; r < 8; ++r) {
            int row = m0 + half * 8 + r;
            float v = acc[r] + bv;
            out [(size_t)row * out_stride + col] = v;
            pred[(size_t)row * F_DIM     + col] = (__bf16)v;
        }
    }
}

extern "C" void kernel_launch(void* const* d_in, const int* in_sizes, int n_in,
                              void* d_out, int out_size, void* d_ws, size_t ws_size,
                              hipStream_t stream)
{
    (void)in_sizes; (void)n_in; (void)out_size;
    const float* inputs = (const float*)d_in[0];
    const float* Wl = (const float*)d_in[1];
    const float* Rl = (const float*)d_in[2];
    const float* bl = (const float*)d_in[3];
    const float* Wc = (const float*)d_in[4];
    const float* Rc = (const float*)d_in[5];
    const float* bc = (const float*)d_in[6];
    const float* Wr = (const float*)d_in[7];
    const float* Rr = (const float*)d_in[8];
    const float* br = (const float*)d_in[9];
    const float* W1 = (const float*)d_in[10];
    const float* b1 = (const float*)d_in[11];
    const float* W2 = (const float*)d_in[12];
    const float* b2 = (const float*)d_in[13];
    float* out = (float*)d_out;

    constexpr int KT1 = U_DIM + F_DIM;   // 1088
    constexpr int KT2 = U_DIM + U_DIM;   // 2048

    // Workspace carve-up (~43 MB total)
    char* ws = (char*)d_ws;
    size_t off = 0;
    auto take = [&](size_t bytes) -> char* {
        char* p = ws + off;
        off += (bytes + 255) & ~(size_t)255;
        return p;
    };
    __bf16* catL = (__bf16*)take((size_t)4 * U_DIM * KT1 * 2);  // [Rl|Wl]^T bf16
    __bf16* catC = (__bf16*)take((size_t)4 * U_DIM * KT1 * 2);  // [Rc|Wc]^T
    __bf16* catR = (__bf16*)take((size_t)4 * U_DIM * KT2 * 2);  // [Rr|Wr]^T
    __bf16* W1T  = (__bf16*)take((size_t)DHID * U_DIM * 2);
    __bf16* W2T  = (__bf16*)take((size_t)F_DIM * DHID * 2);
    __bf16* inb  = (__bf16*)take((size_t)B_DIM * T_DIM * F_DIM * 2);  // inputs bf16
    __bf16* hA   = (__bf16*)take((size_t)B_DIM * U_DIM * 2);
    __bf16* hB   = (__bf16*)take((size_t)B_DIM * U_DIM * 2);
    __bf16* hrA  = (__bf16*)take((size_t)B_DIM * U_DIM * 2);
    __bf16* hrB  = (__bf16*)take((size_t)B_DIM * U_DIM * 2);
    float*  cA   = (float*)take((size_t)B_DIM * U_DIM * 4);
    float*  cB   = (float*)take((size_t)B_DIM * U_DIM * 4);
    float*  crA  = (float*)take((size_t)B_DIM * U_DIM * 4);
    float*  crB  = (float*)take((size_t)B_DIM * U_DIM * 4);
    __bf16* pred = (__bf16*)take((size_t)B_DIM * F_DIM * 2);
    if (off > ws_size) return;   // workspace too small — nothing sane to do

    // --- per-call weight transpose + bf16 downconvert (stays resident in L2) ---
    auto tr = [&](__bf16* dst, int pitch, const float* src, int K, int N) {
        int total = K * N;
        transpose_to_bf16<<<(total + 255) / 256, 256, 0, stream>>>(dst, pitch, src, K, N);
    };
    tr(catL,         KT1, Rl, U_DIM, 4 * U_DIM);
    tr(catL + U_DIM, KT1, Wl, F_DIM, 4 * U_DIM);
    tr(catC,         KT1, Rc, U_DIM, 4 * U_DIM);
    tr(catC + U_DIM, KT1, Wc, F_DIM, 4 * U_DIM);
    tr(catR,         KT2, Rr, U_DIM, 4 * U_DIM);
    tr(catR + U_DIM, KT2, Wr, U_DIM, 4 * U_DIM);
    tr(W1T, U_DIM, W1, U_DIM, DHID);
    tr(W2T, DHID,  W2, DHID,  F_DIM);

    {   // encoder inputs -> bf16 once
        int total = B_DIM * T_DIM * F_DIM;
        cast_to_bf16<<<(total + 255) / 256, 256, 0, stream>>>(inb, inputs, total);
    }
    zero_buf<<<(B_DIM * U_DIM + 255) / 256, 256, 0, stream>>>(cA, hA, B_DIM * U_DIM);

    const size_t sh1 = (size_t)16 * (KT1 + 8) * sizeof(__bf16);  // 35,072 B
    const size_t sh2 = (size_t)16 * (KT2 + 8) * sizeof(__bf16);  // 65,792 B (>64KB: opt in)
    (void)hipFuncSetAttribute((const void*)lstm_step_kernel<KT2>,
                              hipFuncAttributeMaxDynamicSharedMemorySize, (int)sh2);

    dim3 lgrid(U_DIM / 64, B_DIM / 16);   // (16, 8)

    // --- encoder scan over T ---
    __bf16 *hc = hA, *hn = hB;
    float  *cc = cA, *cn = cB;
    for (int t = 0; t < T_DIM; ++t) {
        lstm_step_kernel<KT1><<<lgrid, 256, sh1, stream>>>(
            hc, cc, inb + (size_t)t * F_DIM, T_DIM * F_DIM, catL, bl, hn, cn);
        std::swap(hc, hn); std::swap(cc, cn);
    }

    // pred0 from encoder final state -> d_out step 0
    dense_head_kernel<<<B_DIM / 16, 128, 0, stream>>>(
        hc, W1T, b1, W2T, b2, out + 0 * F_DIM, NOUT * F_DIM, pred);

    // rnn_cell state initialized from encoder final state (evolves separately)
    (void)hipMemcpyAsync(hrA, hc, (size_t)B_DIM * U_DIM * 2, hipMemcpyDeviceToDevice, stream);
    (void)hipMemcpyAsync(crA, cc, (size_t)B_DIM * U_DIM * 4, hipMemcpyDeviceToDevice, stream);
    __bf16 *hrc = hrA, *hrn = hrB;
    float  *crc = crA, *crn = crB;

    // --- autoregressive decode ---
    for (int s = 1; s < NOUT; ++s) {
        // lstm_cell: x = previous prediction [B,F] (bf16)
        lstm_step_kernel<KT1><<<lgrid, 256, sh1, stream>>>(
            hc, cc, pred, F_DIM, catC, bc, hn, cn);
        std::swap(hc, hn); std::swap(cc, cn);
        // rnn_cell: x = lstm_cell output h [B,U] (bf16)
        lstm_step_kernel<KT2><<<lgrid, 256, sh2, stream>>>(
            hrc, crc, hc, U_DIM, catR, br, hrn, crn);
        std::swap(hrc, hrn); std::swap(crc, crn);
        // dense head -> d_out step s (f32) and pred feedback (bf16)
        dense_head_kernel<<<B_DIM / 16, 128, 0, stream>>>(
            hrc, W1T, b1, W2T, b2, out + (size_t)s * F_DIM, NOUT * F_DIM, pred);
    }
}
